// GraphormerLayerSparse_17205638988081
// MI455X (gfx1250) — matrix operand order repaired and checked by reading, not compiled, not run on hardware
//
#include <hip/hip_runtime.h>
#include <cstdint>
#include <cstddef>

#define HID 128
#define NHEAD 8
#define HD 16

typedef __attribute__((ext_vector_type(16))) __bf16 v16bf;
typedef __attribute__((ext_vector_type(8)))  __bf16 v8bf;
typedef __attribute__((ext_vector_type(8)))  float  v8f;

typedef unsigned int uint32x4 __attribute__((ext_vector_type(4)));
typedef int          int32x4  __attribute__((ext_vector_type(4)));
typedef int          int32x8  __attribute__((ext_vector_type(8)));

__device__ __forceinline__ unsigned short f2bf(float f) {
    unsigned u = __float_as_uint(f);
    u += 0x7fffu + ((u >> 16) & 1u);          // round-to-nearest-even
    return (unsigned short)(u >> 16);
}

// ---------------------------------------------------------------- conversions
// weights: f32 [K][Nt] row-major  ->  bf16 [Nt][K] (transposed, K contiguous)
__global__ void k_convT(const float* __restrict__ s,
                        unsigned short* __restrict__ d, int K, int Nt) {
    int i = blockIdx.x * blockDim.x + threadIdx.x;
    if (i < K * Nt) {
        int k = i / Nt, n = i - k * Nt;
        d[(size_t)n * K + k] = f2bf(s[i]);
    }
}

__global__ void k_zero(float* __restrict__ p, size_t n) {
    size_t i = (size_t)blockIdx.x * blockDim.x + threadIdx.x;
    if (i < n) p[i] = 0.f;
}

// ---------------------------------------------------------------- WMMA GEMM
// C[M,Nt] = A[M,K] @ B[K,Nt] + bias.  A: f32 (converted) or bf16 row-major.
// Bt: bf16 TRANSPOSED [Nt][K].  128x128 tile/block, 8 waves, 32x64 per wave.
// B tile is DMA'd into LDS by the Tensor Data Mover when available.
template <bool RELU, bool OUT_BF16, bool A_F32>
__global__ __launch_bounds__(256)
void k_gemm_bias(const void* __restrict__ Ap,
                 const unsigned short* __restrict__ Bt,
                 const float* __restrict__ bias,
                 void* __restrict__ Cp,
                 int M, int Nt, int K) {
    __shared__ __align__(32) unsigned short As[128][32];
    __shared__ __align__(32) unsigned short Bs[128][32];   // [n][k]

    const int m0   = blockIdx.x * 128;
    const int n0   = blockIdx.y * 128;
    const int t    = threadIdx.x;
    const int lane = t & 31;
    const int wave = t >> 5;
    const int wm   = (wave & 3) * 32;   // 4 wave-rows of 32
    const int wn   = (wave >> 2) * 64;  // 2 wave-cols of 64

    v8f acc[2][4] = {};

    const float*          Af = (const float*)Ap;
    const unsigned short* Ab = (const unsigned short*)Ap;

    for (int k0 = 0; k0 < K; k0 += 32) {
        // ---- A tile -> LDS (bf16), 2 threads per row, 16 elems each
        {
            int row = t >> 1;
            int cb  = (t & 1) * 16;
            int gm  = m0 + row;
            if (A_F32) {
#pragma unroll
                for (int i = 0; i < 16; i += 4) {
                    float4 f;
                    if (gm < M) f = *(const float4*)(Af + (size_t)gm * K + k0 + cb + i);
                    else        f = make_float4(0.f, 0.f, 0.f, 0.f);
                    As[row][cb + i + 0] = f2bf(f.x);
                    As[row][cb + i + 1] = f2bf(f.y);
                    As[row][cb + i + 2] = f2bf(f.z);
                    As[row][cb + i + 3] = f2bf(f.w);
                }
            } else {
                if (gm < M) {
                    const uint4* src = (const uint4*)(Ab + (size_t)gm * K + k0 + cb);
                    *(uint4*)&As[row][cb]     = src[0];
                    *(uint4*)&As[row][cb + 8] = src[1];
                } else {
                    uint4 z = {0u, 0u, 0u, 0u};
                    *(uint4*)&As[row][cb]     = z;
                    *(uint4*)&As[row][cb + 8] = z;
                }
            }
            if (k0 + 32 < K && gm < M) {
                if (A_F32) __builtin_prefetch(Af + (size_t)gm * K + k0 + 32 + cb, 0, 0);
                else       __builtin_prefetch(Ab + (size_t)gm * K + k0 + 32 + cb, 0, 0);
            }
        }

        // ---- B tile -> LDS: TDM 2D tile DMA (128 rows x 32 bf16, stride K)
#if __has_builtin(__builtin_amdgcn_tensor_load_to_lds)
        if (t < 32) {   // wave 0 issues the DMA; TDM ignores EXEC
            unsigned long long ga =
                (unsigned long long)(const char*)(Bt + ((size_t)n0 * K + k0) * 1) * 1ull;
            ga = (unsigned long long)(uintptr_t)(Bt + (size_t)n0 * K + k0);
            unsigned lds_b = (unsigned)(uintptr_t)&Bs[0][0];
            // D# group0 (ISA 8.3): count=1 | lds_addr | global_addr | type=2
            uint32x4 g0;
            g0[0] = 1u;
            g0[1] = lds_b;
            g0[2] = (unsigned)(ga & 0xffffffffu);
            g0[3] = (unsigned)((ga >> 32) & 0x1ffffffu) | (2u << 30);
            // D# group1 (ISA 8.4): data_size=2B, dims/tile/stride
            unsigned uK = (unsigned)K, uN = (unsigned)Nt;
            int32x8 g1;
            g1[0] = (int)(1u << 16);                               // data_size=1 (2B)
            g1[1] = (int)((uK & 0xffffu) << 16);                   // tensor_dim0[15:0]
            g1[2] = (int)((uK >> 16) | ((uN & 0xffffu) << 16));    // dim0[31:16]|dim1[15:0]
            g1[3] = (int)((uN >> 16) | (32u << 16));               // dim1[31:16]|tile_dim0=32
            g1[4] = (int)128u;                                     // tile_dim1=128, tile_dim2=0
            g1[5] = (int)uK;                                       // dim0_stride[31:0]
            g1[6] = 0;                                             // dim0_stride[47:32]
            g1[7] = 0;
            int32x4 gz = {0, 0, 0, 0};
#if defined(__clang_major__) && (__clang_major__ >= 23)
            int32x8 gz8 = {0, 0, 0, 0, 0, 0, 0, 0};
            __builtin_amdgcn_tensor_load_to_lds(g0, g1, gz, gz, gz8, 0);
#else
            __builtin_amdgcn_tensor_load_to_lds(g0, g1, gz, gz, 0);
#endif
            __builtin_amdgcn_s_wait_tensorcnt(0);
        }
#else
        // fallback: contiguous vector loads from transposed B
        {
            int row = t >> 1;
            int cb  = (t & 1) * 16;
            const uint4* src = (const uint4*)(Bt + (size_t)(n0 + row) * K + k0 + cb);
            *(uint4*)&Bs[row][cb]     = src[0];
            *(uint4*)&Bs[row][cb + 8] = src[1];
            if (k0 + 32 < K)
                __builtin_prefetch(Bt + (size_t)(n0 + row) * K + k0 + 32 + cb, 0, 0);
        }
#endif
        __syncthreads();

        // ---- fragments (ISA 7.12.2 16-bit layouts, wave32)
        v16bf af[2];
#pragma unroll
        for (int i = 0; i < 2; ++i) {
            int m  = wm + i * 16 + (lane & 15);
            int kb = (lane < 16) ? 0 : 8;        // A: K kb..kb+7 and 16+kb..16+kb+7
            union { v16bf v; v8bf h[2]; } u;
            u.h[0] = *(const v8bf*)&As[m][kb];
            u.h[1] = *(const v8bf*)&As[m][16 + kb];
            af[i] = u.v;
        }
        v16bf bf[4];
#pragma unroll
        for (int j = 0; j < 4; ++j) {
            int n  = wn + j * 16 + (lane & 15);
            int ks = (lane < 16) ? 0 : 8;        // B: K ks..ks+7 and 16+ks..16+ks+7
            union { v16bf v; v8bf h[2]; } u;
            u.h[0] = *(const v8bf*)&Bs[n][ks];
            u.h[1] = *(const v8bf*)&Bs[n][16 + ks];
            bf[j] = u.v;
        }
#pragma unroll
        for (int i = 0; i < 2; ++i)
#pragma unroll
            for (int j = 0; j < 4; ++j)
                acc[i][j] = __builtin_amdgcn_wmma_f32_16x16x32_bf16(
                    false, af[i], false, bf[j], (short)0, acc[i][j], false, false);
        __syncthreads();
    }

    // ---- epilogue: C/D layout: lane&15 = N col, VGPR r -> M = r (+8 hi half)
    const int ncol = lane & 15;
    const int rofs = (lane < 16) ? 0 : 8;
#pragma unroll
    for (int i = 0; i < 2; ++i) {
#pragma unroll
        for (int j = 0; j < 4; ++j) {
            int   gn = n0 + wn + j * 16 + ncol;
            float bv = bias[gn];
#pragma unroll
            for (int r = 0; r < 8; ++r) {
                int gm = m0 + wm + i * 16 + rofs + r;
                if (gm < M) {
                    float val = acc[i][j][r] + bv;
                    if (RELU) val = fmaxf(val, 0.f);
                    if (OUT_BF16)
                        ((unsigned short*)Cp)[(size_t)gm * Nt + gn] = f2bf(val);
                    else
                        ((float*)Cp)[(size_t)gm * Nt + gn] = val;
                }
            }
        }
    }
}

// ---------------------------------------------------------------- edge stage
__global__ void k_edge_score(const long long* __restrict__ ei,
                             const float* __restrict__ q,
                             const float* __restrict__ k,
                             const float* __restrict__ ea,
                             const float* __restrict__ We,
                             const float* __restrict__ be,
                             float* __restrict__ score, int E) {
    long long gid = (long long)blockIdx.x * blockDim.x + threadIdx.x;
    if (gid >= (long long)E * NHEAD) return;
    int e = (int)(gid >> 3), h = (int)(gid & 7);
    long long s = ei[e], d = ei[E + e];
    const float* qr = q + (size_t)s * HID + h * HD;
    const float* kr = k + (size_t)d * HID + h * HD;
    float acc = 0.f;
#pragma unroll
    for (int i = 0; i < HD; ++i) acc += qr[i] * kr[i];
    acc *= 0.25f;                       // 1/sqrt(16)
    const float* er = ea + (size_t)e * 16;
    float eb = be[h];
#pragma unroll
    for (int i = 0; i < 16; ++i) eb += er[i] * We[i * NHEAD + h];
    score[(size_t)e * NHEAD + h] = acc + eb;
}

// per-head block-partial max over all edges (i % 8 constant per thread)
__global__ void k_score_max_partial(const float* __restrict__ score,
                                    float* __restrict__ partial, size_t total) {
    float m = -3.4e38f;
    for (size_t i = (size_t)blockIdx.x * blockDim.x + threadIdx.x; i < total;
         i += (size_t)gridDim.x * blockDim.x)
        m = fmaxf(m, score[i]);
    __shared__ float sm[256];
    sm[threadIdx.x] = m;
    __syncthreads();
    for (int off = 128; off >= 8; off >>= 1) {
        if (threadIdx.x < (unsigned)off)
            sm[threadIdx.x] = fmaxf(sm[threadIdx.x], sm[threadIdx.x + off]);
        __syncthreads();
    }
    if (threadIdx.x < 8) partial[blockIdx.x * 8 + threadIdx.x] = sm[threadIdx.x];
}

__global__ void k_score_sum_partial(const float* __restrict__ score,
                                    const float* __restrict__ maxv,
                                    float* __restrict__ partial, size_t total) {
    float mv = maxv[threadIdx.x & 7];
    float s  = 0.f;
    for (size_t i = (size_t)blockIdx.x * blockDim.x + threadIdx.x; i < total;
         i += (size_t)gridDim.x * blockDim.x)
        s += __expf(score[i] - mv);
    __shared__ float sm[256];
    sm[threadIdx.x] = s;
    __syncthreads();
    for (int off = 128; off >= 8; off >>= 1) {
        if (threadIdx.x < (unsigned)off)
            sm[threadIdx.x] += sm[threadIdx.x + off];
        __syncthreads();
    }
    if (threadIdx.x < 8) partial[blockIdx.x * 8 + threadIdx.x] = sm[threadIdx.x];
}

__global__ void k_reduce8(const float* __restrict__ partial,
                          float* __restrict__ out8, int G, int isSum) {
    int h = threadIdx.x;
    if (h >= 8) return;
    if (isSum) {
        float s = 0.f;
        for (int b = 0; b < G; ++b) s += partial[b * 8 + h];
        out8[h] = s;
    } else {
        float m = -3.4e38f;
        for (int b = 0; b < G; ++b) m = fmaxf(m, partial[b * 8 + h]);
        out8[h] = m;
    }
}

__global__ void k_edge_message(const long long* __restrict__ ei,
                               const float* __restrict__ v,
                               const float* __restrict__ score,
                               const float* __restrict__ maxv,
                               const float* __restrict__ sumv,
                               float* __restrict__ agg, int E) {
    long long gid = (long long)blockIdx.x * blockDim.x + threadIdx.x;
    if (gid >= (long long)E * HID) return;
    int e = (int)(gid >> 7), c = (int)(gid & 127), h = c >> 4;
    long long s = ei[e], d = ei[E + e];
    float w = __expf(score[(size_t)e * NHEAD + h] - maxv[h]) / sumv[h];
    atomicAdd(&agg[(size_t)s * HID + c], w * v[(size_t)d * HID + c]);
}

// ---------------------------------------------------------------- layernorm
__global__ void k_ln_add(const float* __restrict__ x, const float* __restrict__ y,
                         const float* __restrict__ g, const float* __restrict__ b,
                         float* __restrict__ out, int M) {
    int row  = blockIdx.x * 8 + (threadIdx.x >> 5);
    int lane = threadIdx.x & 31;
    if (row >= M) return;
    const float* xr = x + (size_t)row * HID;
    const float* yr = y + (size_t)row * HID;
    float vals[4], s = 0.f;
#pragma unroll
    for (int i = 0; i < 4; ++i) {
        vals[i] = xr[lane + 32 * i] + yr[lane + 32 * i];
        s += vals[i];
    }
    for (int off = 16; off; off >>= 1) s += __shfl_xor(s, off, 32);
    float mu = s * (1.f / 128.f);
    float var = 0.f;
#pragma unroll
    for (int i = 0; i < 4; ++i) {
        float d = vals[i] - mu;
        var += d * d;
    }
    for (int off = 16; off; off >>= 1) var += __shfl_xor(var, off, 32);
    float r = rsqrtf(var * (1.f / 128.f) + 1e-5f);
#pragma unroll
    for (int i = 0; i < 4; ++i) {
        int c = lane + 32 * i;
        out[(size_t)row * HID + c] = (vals[i] - mu) * r * g[c] + b[c];
    }
}

// ---------------------------------------------------------------- launcher
extern "C" void kernel_launch(void* const* d_in, const int* in_sizes, int n_in,
                              void* d_out, int out_size, void* d_ws, size_t ws_size,
                              hipStream_t stream) {
    (void)n_in; (void)out_size; (void)ws_size;
    const float*     x    = (const float*)d_in[0];
    const long long* ei   = (const long long*)d_in[1];
    const float*     ea   = (const float*)d_in[2];
    const float*     Wq   = (const float*)d_in[3];
    const float*     bq   = (const float*)d_in[4];
    const float*     Wk   = (const float*)d_in[5];
    const float*     bk   = (const float*)d_in[6];
    const float*     Wv   = (const float*)d_in[7];
    const float*     bv   = (const float*)d_in[8];
    const float*     Wo   = (const float*)d_in[9];
    const float*     bo   = (const float*)d_in[10];
    const float*     We   = (const float*)d_in[11];
    const float*     be   = (const float*)d_in[12];
    const float*     g1   = (const float*)d_in[13];
    const float*     b1   = (const float*)d_in[14];
    const float*     g2   = (const float*)d_in[15];
    const float*     b2   = (const float*)d_in[16];
    const float*     Wff1 = (const float*)d_in[17];
    const float*     bff1 = (const float*)d_in[18];
    const float*     Wff2 = (const float*)d_in[19];
    const float*     bff2 = (const float*)d_in[20];

    const int N = in_sizes[0] / HID;
    const int E = in_sizes[1] / 2;

    char*  base = (char*)d_ws;
    size_t off  = 0;
    auto alloc = [&](size_t bytes) -> char* {
        char* p = base + off;
        off += (bytes + 255) & ~(size_t)255;
        return p;
    };
    unsigned short* wb      = (unsigned short*)alloc(196608u * 2);
    float*          qb      = (float*)alloc((size_t)N * HID * 4);
    float*          kbuf    = (float*)alloc((size_t)N * HID * 4);
    float*          vbuf    = (float*)alloc((size_t)N * HID * 4);
    float*          score   = (float*)alloc((size_t)E * NHEAD * 4);
    float*          partial = (float*)alloc(512u * 8 * 4);
    float*          maxv    = (float*)alloc(256);
    float*          sumv    = (float*)alloc(256);
    float*          agg     = (float*)alloc((size_t)N * HID * 4);
    float*          h       = (float*)alloc((size_t)N * HID * 4);
    unsigned short* ffmid   = (unsigned short*)alloc((size_t)N * 512 * 2);
    float*          attno   = qb;     // reuse: q dead after score kernel
    float*          ffout   = kbuf;   // reuse: k dead after score kernel

    unsigned short* wq_b  = wb;              // all stored transposed [Nt][K]
    unsigned short* wk_b  = wb + 16384;
    unsigned short* wv_b  = wb + 32768;
    unsigned short* wo_b  = wb + 49152;
    unsigned short* wf1_b = wb + 65536;
    unsigned short* wf2_b = wb + 131072;

    k_convT<<<64, 256, 0, stream>>>(Wq, wq_b, 128, 128);
    k_convT<<<64, 256, 0, stream>>>(Wk, wk_b, 128, 128);
    k_convT<<<64, 256, 0, stream>>>(Wv, wv_b, 128, 128);
    k_convT<<<64, 256, 0, stream>>>(Wo, wo_b, 128, 128);
    k_convT<<<256, 256, 0, stream>>>(Wff1, wf1_b, 128, 512);
    k_convT<<<256, 256, 0, stream>>>(Wff2, wf2_b, 512, 128);

    size_t aggN = (size_t)N * HID;
    k_zero<<<(int)((aggN + 255) / 256), 256, 0, stream>>>(agg, aggN);

    dim3 blk(256);
    int  mb = (N + 127) / 128;
    k_gemm_bias<false, false, true><<<dim3(mb, 1), blk, 0, stream>>>(x, wq_b, bq, qb,   N, 128, 128);
    k_gemm_bias<false, false, true><<<dim3(mb, 1), blk, 0, stream>>>(x, wk_b, bk, kbuf, N, 128, 128);
    k_gemm_bias<false, false, true><<<dim3(mb, 1), blk, 0, stream>>>(x, wv_b, bv, vbuf, N, 128, 128);

    long long totalEH = (long long)E * NHEAD;
    k_edge_score<<<(int)((totalEH + 255) / 256), 256, 0, stream>>>(ei, qb, kbuf, ea, We, be, score, E);

    k_score_max_partial<<<512, 256, 0, stream>>>(score, partial, (size_t)E * NHEAD);
    k_reduce8<<<1, 32, 0, stream>>>(partial, maxv, 512, 0);
    k_score_sum_partial<<<512, 256, 0, stream>>>(score, maxv, partial, (size_t)E * NHEAD);
    k_reduce8<<<1, 32, 0, stream>>>(partial, sumv, 512, 1);

    long long totalEC = (long long)E * HID;
    k_edge_message<<<(int)((totalEC + 255) / 256), 256, 0, stream>>>(ei, vbuf, score, maxv, sumv, agg, E);

    k_gemm_bias<false, false, true><<<dim3(mb, 1), blk, 0, stream>>>(agg, wo_b, bo, attno, N, 128, 128);

    k_ln_add<<<(N + 7) / 8, 256, 0, stream>>>(x, attno, g1, b1, h, N);

    k_gemm_bias<true, true, true><<<dim3(mb, 4), blk, 0, stream>>>(h, wf1_b, bff1, ffmid, N, 512, 128);
    k_gemm_bias<false, false, false><<<dim3(mb, 1), blk, 0, stream>>>(ffmid, wf2_b, bff2, ffout, N, 128, 512);

    k_ln_add<<<(N + 7) / 8, 256, 0, stream>>>(h, ffout, g2, b2, (float*)d_out, N);
}
